// SFAANetwork_88399016886454
// MI455X (gfx1250) — matrix operand order repaired and checked by reading, not compiled, not run on hardware
//
#include <hip/hip_runtime.h>

// ---------------------------------------------------------------------------
// Block-sparse quantized-KV attention for gfx1250 (MI455X).
//   B=2, N1=32, N2=8, S1=2048, S2=8192, D=128, BS=16, KSEL=64, G=4
// One workgroup (4 waves) = one (b, kv-head, query-block); wave w handles
// query head h1 = n2*4 + w. K/V blocks are gathered+dequant-converted to f16
// in LDS once and consumed by all 4 waves via v_wmma_f32_16x16x32_f16.
// ---------------------------------------------------------------------------

typedef __attribute__((ext_vector_type(16))) _Float16 v16h;
typedef __attribute__((ext_vector_type(8)))  float    v8f;

union F16x16 { uint4 q[2]; _Float16 h[16]; v16h v; };

constexpr int B_    = 2;
constexpr int N1_   = 32;
constexpr int N2_   = 8;
constexpr int S1_   = 2048;
constexpr int S2_   = 8192;
constexpr int D_    = 128;
constexpr int BS_   = 16;
constexpr int QB_   = S1_ / BS_;   // 128
constexpr int KVB_  = S2_ / BS_;   // 512
constexpr int KSEL_ = 64;
constexpr int G_    = N1_ / N2_;   // 4
constexpr int PAIRS = KSEL_ / 2;   // 32

__global__ __launch_bounds__(128)
void sfaa_attn_kernel(const float* __restrict__ qry,
                      const int*   __restrict__ keyq,
                      const int*   __restrict__ valq,
                      const int*   __restrict__ sidx,
                      const float* __restrict__ kds,
                      const float* __restrict__ vds,
                      const float* __restrict__ svp,
                      float*       __restrict__ out)
{
    // K pair-block, row-major f16, rows padded to 136 halfs (17 uint4, 16B align)
    __shared__ uint4 KbufV[32][17];
    // V pair-block, d-major (transposed), rows padded to 40 halfs (5 uint4)
    __shared__ uint4 VtV[128][5];
    // per-wave raw scores 16x32 (rows padded to 33 floats)
    __shared__ float Sbuf[4][16 * 33];
    // per-wave P (exp * vscale) 16x32 f16, rows padded to 40 halfs
    __shared__ uint4 PbufV[4][16][5];
    __shared__ float rBuf[4][16];   // per-row rescale factor this pair
    __shared__ float lBuf[4][16];   // per-row running softmax denominator

    const int tid  = threadIdx.x;
    const int w    = tid >> 5;        // wave id == g (query head in group)
    const int lane = tid & 31;
    const bool lo  = lane < 16;
    const int nc   = lane & 15;       // N-column (B/C/D) or M-row (A)
    const int mo   = lo ? 0 : 8;      // C/D row offset for this half-wave

    const int qb = blockIdx.x % QB_;
    const int n2 = (blockIdx.x / QB_) % N2_;
    const int b  = blockIdx.x / (QB_ * N2_);
    const int h1 = n2 * G_ + w;

    const float sv = svp[0];
    const int*   idx = sidx + (((long)(b * N2_ + n2)) * QB_ + qb) * KSEL_;
    const float* ksc = kds + (long)(b * N2_ + n2) * KVB_;
    const float* vsc = vds + (long)(b * N2_ + n2) * KVB_;
    const long kvHeadBase = ((long)(b * N2_ + n2)) * S2_ * D_;
    const long qRowBase   = (((long)(b * N1_ + h1)) * S1_ + (long)qb * BS_) * D_;

    // ---- Q tile (16x128) -> 4 A-fragments (16x32 f16), ISA A layout ----
    v16h qf[4];
    {
        const float4* q4 = reinterpret_cast<const float4*>(qry + qRowBase + (long)nc * D_);
        #pragma unroll
        for (int kk = 0; kk < 4; ++kk) {
            const int base = kk * 8 + (lo ? 0 : 2);          // float4 units
            float4 a0 = q4[base],     a1 = q4[base + 1];     // K = kk*32 + {0..7 | 8..15}
            float4 b0 = q4[base + 4], b1 = q4[base + 5];     // K = kk*32 + {16..23 | 24..31}
            v16h f;
            f[0]=(_Float16)a0.x; f[1]=(_Float16)a0.y; f[2]=(_Float16)a0.z; f[3]=(_Float16)a0.w;
            f[4]=(_Float16)a1.x; f[5]=(_Float16)a1.y; f[6]=(_Float16)a1.z; f[7]=(_Float16)a1.w;
            f[8]=(_Float16)b0.x; f[9]=(_Float16)b0.y; f[10]=(_Float16)b0.z; f[11]=(_Float16)b0.w;
            f[12]=(_Float16)b1.x; f[13]=(_Float16)b1.y; f[14]=(_Float16)b1.z; f[15]=(_Float16)b1.w;
            qf[kk] = f;
        }
    }

    v8f acc[8];
    {
        v8f z = {};
        #pragma unroll
        for (int nt = 0; nt < 8; ++nt) acc[nt] = z;
    }

    float mrun = -INFINITY;   // meaningful in lanes 0..15 only
    float lrun = 0.0f;

    float* Sw = Sbuf[w];
    float* rW = rBuf[w];
    float* lW = lBuf[w];

    for (int it = 0; it < PAIRS; ++it) {
        const int j0 = idx[2 * it], j1 = idx[2 * it + 1];
        const float ks0 = ksc[j0] * sv, ks1 = ksc[j1] * sv;
        const float vs0 = vsc[j0],      vs1 = vsc[j1];

        __syncthreads();   // previous pair's LDS reads complete before refill

        // ---- gather + int->f16 convert: K row-major, V transposed ----
        {
            const int r    = tid & 31;        // pair-local kv row (0..31)
            const int cseg = tid >> 5;        // 32-col segment (0..3)
            const int blk  = (r < 16) ? j0 : j1;
            const long g0  = kvHeadBase + (long)(blk * BS_ + (r & 15)) * D_ + cseg * 32;
            const int4* kp = reinterpret_cast<const int4*>(keyq + g0);
            const int4* vp = reinterpret_cast<const int4*>(valq + g0);
            _Float16* krow = reinterpret_cast<_Float16*>(&KbufV[r][0]) + cseg * 32;
            _Float16* vt0  = reinterpret_cast<_Float16*>(&VtV[0][0]);
            #pragma unroll
            for (int u = 0; u < 8; ++u) {
                int4 kv4 = kp[u];
                int4 vv4 = vp[u];
                const int c = cseg * 32 + u * 4;
                krow[u*4+0] = (_Float16)(float)kv4.x;
                krow[u*4+1] = (_Float16)(float)kv4.y;
                krow[u*4+2] = (_Float16)(float)kv4.z;
                krow[u*4+3] = (_Float16)(float)kv4.w;
                vt0[(c+0)*40 + r] = (_Float16)(float)vv4.x;
                vt0[(c+1)*40 + r] = (_Float16)(float)vv4.y;
                vt0[(c+2)*40 + r] = (_Float16)(float)vv4.z;
                vt0[(c+3)*40 + r] = (_Float16)(float)vv4.w;
            }
            if (it + 1 < PAIRS) {     // global_prefetch_b8 next pair's rows
                const int nblk = (r < 16) ? idx[2*it+2] : idx[2*it+3];
                const long g1 = kvHeadBase + (long)(nblk * BS_ + (r & 15)) * D_ + cseg * 32;
                __builtin_prefetch((const void*)(keyq + g1), 0, 0);
                __builtin_prefetch((const void*)(valq + g1), 0, 0);
            }
        }
        __syncthreads();

        // ---- S = Q . K^T : two 16x16 N-tiles (one per kv block), K=128 ----
        v8f s01[2];
        #pragma unroll
        for (int jj = 0; jj < 2; ++jj) {
            v8f s = {};
            const uint4* krow = &KbufV[jj * 16 + nc][0];
            #pragma unroll
            for (int kk = 0; kk < 4; ++kk) {
                F16x16 t;
                const int bi = kk * 4 + (lo ? 0 : 1);   // K = kk*32 + {0..7,16..23 | 8..15,24..31}
                t.q[0] = krow[bi];
                t.q[1] = krow[bi + 2];
                s = __builtin_amdgcn_wmma_f32_16x16x32_f16(
                        false, qf[kk], false, t.v, (short)0, s, false, false);
            }
            s01[jj] = s;
        }

        // ---- spill raw scores to per-wave LDS (D layout -> row major) ----
        #pragma unroll
        for (int jj = 0; jj < 2; ++jj)
            #pragma unroll
            for (int r = 0; r < 8; ++r)
                Sw[(r + mo) * 33 + jj * 16 + nc] = s01[jj][r];

        __syncthreads();

        // ---- online softmax row stats: lane r < 16 owns score row r ----
        if (lo) {
            const float* srow = Sw + lane * 33;
            float tmp[32];
            float pm = -INFINITY;
            #pragma unroll
            for (int c = 0; c < 32; ++c) {
                float v = srow[c] * (c < 16 ? ks0 : ks1);
                tmp[c] = v;
                pm = fmaxf(pm, v);
            }
            const float mnew = fmaxf(mrun, pm);
            const float resc = __expf(mrun - mnew);    // exp(-inf)=0 on first pair
            float lsum = 0.0f;
            _Float16* prow = reinterpret_cast<_Float16*>(&PbufV[w][lane][0]);
            #pragma unroll
            for (int c = 0; c < 32; ++c) {
                float e = __expf(tmp[c] - mnew);
                lsum += e;
                prow[c] = (_Float16)(e * (c < 16 ? vs0 : vs1)); // fold V scale into P
            }
            lrun = lrun * resc + lsum;
            mrun = mnew;
            rW[lane] = resc;
            lW[lane] = lrun;
        }
        __syncthreads();

        // ---- rescale accumulators by per-row factor ----
        float fr[8];
        #pragma unroll
        for (int r = 0; r < 8; ++r) fr[r] = rW[r + mo];
        #pragma unroll
        for (int nt = 0; nt < 8; ++nt)
            #pragma unroll
            for (int r = 0; r < 8; ++r) acc[nt][r] *= fr[r];

        // ---- P A-fragment (16x32 f16, ISA A layout) ----
        F16x16 pa;
        {
            const uint4* prow = &PbufV[w][nc][0];
            pa.q[0] = prow[lo ? 0 : 1];   // K = {0..7 | 8..15}
            pa.q[1] = prow[lo ? 2 : 3];   // K = {16..23 | 24..31}
        }

        // ---- O += P . V : 8 N-tiles over D, K = 32 kv rows ----
        #pragma unroll
        for (int nt = 0; nt < 8; ++nt) {
            const uint4* vrow = &VtV[nt * 16 + nc][0];
            F16x16 t;
            t.q[0] = vrow[lo ? 0 : 1];    // K = {0..7 | 8..15}
            t.q[1] = vrow[lo ? 2 : 3];    // K = {16..23 | 24..31}
            acc[nt] = __builtin_amdgcn_wmma_f32_16x16x32_f16(
                          false, pa.v, false, t.v, (short)0, acc[nt], false, false);
        }
    }

    // ---- epilogue: divide by softmax denominator, write out ----
    float linv[8];
    #pragma unroll
    for (int r = 0; r < 8; ++r) linv[r] = 1.0f / lW[r + mo];
    #pragma unroll
    for (int nt = 0; nt < 8; ++nt)
        #pragma unroll
        for (int r = 0; r < 8; ++r)
            out[qRowBase + (long)(r + mo) * D_ + nt * 16 + nc] = acc[nt][r] * linv[r];
}

extern "C" void kernel_launch(void* const* d_in, const int* in_sizes, int n_in,
                              void* d_out, int out_size, void* d_ws, size_t ws_size,
                              hipStream_t stream) {
    const float* qry  = (const float*)d_in[0];
    const int*   keyq = (const int*)  d_in[1];
    const int*   valq = (const int*)  d_in[2];
    const int*   sidx = (const int*)  d_in[3];
    const float* kds  = (const float*)d_in[4];
    const float* vds  = (const float*)d_in[5];
    const float* svp  = (const float*)d_in[6];
    float* outp = (float*)d_out;

    dim3 grid(B_ * N2_ * QB_);   // 2048 workgroups
    dim3 block(128);             // 4 wave32, one per query head in the group
    sfaa_attn_kernel<<<grid, block, 0, stream>>>(qry, keyq, valq, sidx, kds, vds, svp, outp);
}
